// MultiHeadAttention_81673098101666
// MI455X (gfx1250) — matrix operand order repaired and checked by reading, not compiled, not run on hardware
//
#include <hip/hip_runtime.h>
#include <hip/hip_bf16.h>

// ---------------------------------------------------------------------------
// MultiHeadAttention for MI455X (gfx1250, wave32, WMMA).
// seq[4,2048,128] fp32 -> out[4,2048,128] fp32
// All GEMMs via v_wmma_f32_16x16x32_f16 (f16 in, f32 accumulate),
// software-pipelined 32x64 wave tiles (double-buffered fragments) with
// WGP-scope global_prefetch look-ahead; LDS-tiled V transpose.
// ---------------------------------------------------------------------------

#define BATCH 4
#define SEQ   2048
#define DMODEL 128
#define DATT  1024
#define M_ALL (BATCH * SEQ)        // 8192
#define QKV_LD (3 * DATT)          // 3072

typedef __attribute__((ext_vector_type(16))) _Float16 v16h;
typedef __attribute__((ext_vector_type(8)))  float    v8f;

union FragU { v16h v; uint4 q[2]; };

// A fragment: 16x32 (MxK), row-major source, lda in halfs (multiple of 8).
// Lanes 0-15: row m0+lane, K=[k0..k0+7],[k0+16..k0+23];
// lanes 16-31: same rows, K=[k0+8..k0+15],[k0+24..k0+31].
__device__ __forceinline__ v16h load_a_frag(const _Float16* __restrict__ A,
                                            int lda, int m0, int k0, int lane) {
  const int m  = m0 + (lane & 15);
  const int kb = k0 + ((lane >> 4) << 3);
  const _Float16* p = A + (size_t)m * lda + kb;
  FragU f;
  f.q[0] = *(const uint4*)(p);        // K = kb .. kb+7
  f.q[1] = *(const uint4*)(p + 16);   // K = kb+16 .. kb+23
  return f.v;
}

// B fragment: 32x16 (KxN) with B[k][n] = Wr[n][k] (Wr row-major, ldw halfs).
// Lane holds col n = n0+(lane&15), K = k0+8*(lane>=16)..+7, then +16..+23.
__device__ __forceinline__ v16h load_b_frag(const _Float16* __restrict__ Wr,
                                            int ldw, int n0, int k0, int lane) {
  const int n  = n0 + (lane & 15);
  const int kb = k0 + ((lane >> 4) << 3);
  const _Float16* p = Wr + (size_t)n * ldw + kb;
  FragU f;
  f.q[0] = *(const uint4*)(p);
  f.q[1] = *(const uint4*)(p + 16);
  return f.v;
}

// C/D 16x16 f32: element r of lane -> row m0 + 8*(lane>=16) + r, col n0+(lane&15)
__device__ __forceinline__ void store_d_f16(_Float16* __restrict__ C, int ldc,
                                            int m0, int n0, v8f acc, float scale,
                                            int lane) {
  const int n  = n0 + (lane & 15);
  const int mb = m0 + ((lane >> 4) << 3);
#pragma unroll
  for (int r = 0; r < 8; ++r)
    C[(size_t)(mb + r) * ldc + n] = (_Float16)(acc[r] * scale);
}

__device__ __forceinline__ void store_d_f32(float* __restrict__ C, int ldc,
                                            int m0, int n0, v8f acc, int lane) {
  const int n  = n0 + (lane & 15);
  const int mb = m0 + ((lane >> 4) << 3);
#pragma unroll
  for (int r = 0; r < 8; ++r)
    C[(size_t)(mb + r) * ldc + n] = acc[r];
}

__device__ __forceinline__ v8f wmma_f16(v16h a, v16h b, v8f c) {
  return __builtin_amdgcn_wmma_f32_16x16x32_f16(
      /*neg_a=*/false, a, /*neg_b=*/false, b,
      /*c_mod=*/(short)0, c, /*reuse_a=*/false, /*reuse_b=*/false);
}

// 32x64 wave tile, software pipelined over KSTEPS K-chunks of 32.
// Next-iteration fragments are fetched before the current WMMAs, and the
// K-step after that is prefetched into the WGP cache (locality 3 ->
// SCOPE_WGP global_prefetch_b8) so the wave stays matrix-throughput-bound
// even on cold L2.
template <int KSTEPS>
__device__ __forceinline__ void wave_gemm_32x64(
    const _Float16* __restrict__ A, int lda, int m0,
    const _Float16* __restrict__ Bw, int ldb, int n0,
    v8f accLo[4], v8f accHi[4], int lane) {
  v16h a0 = load_a_frag(A, lda, m0, 0, lane);
  v16h a1 = load_a_frag(A, lda, m0 + 16, 0, lane);
  v16h b[4];
#pragma unroll
  for (int t = 0; t < 4; ++t) b[t] = load_b_frag(Bw, ldb, n0 + 16 * t, 0, lane);

  for (int ks = 0; ks < KSTEPS - 1; ++ks) {
    const int kn = (ks + 1) * 32;
    // Prefetch next K-step fragments (issue loads before the WMMAs below).
    v16h a0n = load_a_frag(A, lda, m0, kn, lane);
    v16h a1n = load_a_frag(A, lda, m0 + 16, kn, lane);
    v16h bn[4];
#pragma unroll
    for (int t = 0; t < 4; ++t) bn[t] = load_b_frag(Bw, ldb, n0 + 16 * t, kn, lane);
    // Cacheline look-ahead two K-steps out, into the WGP-level cache.
    if (ks + 2 < KSTEPS) {
      const int kp = (ks + 2) * 32;
      __builtin_prefetch(A + (size_t)(m0 + (lane & 31)) * lda + kp, 0, 3);
      __builtin_prefetch(Bw + (size_t)(n0 + (lane & 31)) * ldb + kp, 0, 3);
    }
    // 8 back-to-back WMMAs on the current buffers.
#pragma unroll
    for (int t = 0; t < 4; ++t) accLo[t] = wmma_f16(a0, b[t], accLo[t]);
#pragma unroll
    for (int t = 0; t < 4; ++t) accHi[t] = wmma_f16(a1, b[t], accHi[t]);
    a0 = a0n; a1 = a1n;
#pragma unroll
    for (int t = 0; t < 4; ++t) b[t] = bn[t];
  }
  // Epilogue: last K-step.
#pragma unroll
  for (int t = 0; t < 4; ++t) accLo[t] = wmma_f16(a0, b[t], accLo[t]);
#pragma unroll
  for (int t = 0; t < 4; ++t) accHi[t] = wmma_f16(a1, b[t], accHi[t]);
}

// ---------------------------------------------------------------------------
// Elementwise fp32 -> f16
__global__ void f32_to_f16_kernel(const float* __restrict__ src,
                                  _Float16* __restrict__ dst, int n) {
  int i = blockIdx.x * 256 + threadIdx.x;
  if (i < n) dst[i] = (_Float16)src[i];
}

// QKV projection: X[8192x128] * Wqkv^T -> QKV[8192x3072] (f16).
// Q columns (n < 1024) pre-scaled by 1/sqrt(1024) = 1/32.
__global__ void qkv_gemm_kernel(const _Float16* __restrict__ X,
                                const _Float16* __restrict__ Wq,
                                _Float16* __restrict__ QKV) {
  const int lane = threadIdx.x & 31;
  const int wave = threadIdx.x >> 5;
  const int m0 = blockIdx.x * 256 + wave * 32;
  const int n0 = blockIdx.y * 64;
  v8f accLo[4] = {}, accHi[4] = {};
  wave_gemm_32x64<DMODEL / 32>(X, DMODEL, m0, Wq, DMODEL, n0, accLo, accHi, lane);
  const float scale = (n0 < DATT) ? 0.03125f : 1.0f;
#pragma unroll
  for (int t = 0; t < 4; ++t) {
    store_d_f16(QKV, QKV_LD, m0, n0 + 16 * t, accLo[t], scale, lane);
    store_d_f16(QKV, QKV_LD, m0 + 16, n0 + 16 * t, accHi[t], scale, lane);
  }
}

// Scores (one batch): S[2048x2048] = Qb * Kb^T (Q pre-scaled).
__global__ void scores_gemm_kernel(const _Float16* __restrict__ Q,
                                   const _Float16* __restrict__ Kr,
                                   float* __restrict__ S) {
  const int lane = threadIdx.x & 31;
  const int wave = threadIdx.x >> 5;
  const int m0 = blockIdx.x * 256 + wave * 32;
  const int n0 = blockIdx.y * 64;
  v8f accLo[4] = {}, accHi[4] = {};
  wave_gemm_32x64<DATT / 32>(Q, QKV_LD, m0, Kr, QKV_LD, n0, accLo, accHi, lane);
#pragma unroll
  for (int t = 0; t < 4; ++t) {
    store_d_f32(S, SEQ, m0, n0 + 16 * t, accLo[t], lane);
    store_d_f32(S, SEQ, m0 + 16, n0 + 16 * t, accHi[t], lane);
  }
}

// Row softmax over 2048 f32 -> f16 probabilities. One block per row.
__global__ void softmax_rows_kernel(const float* __restrict__ S,
                                    _Float16* __restrict__ P) {
  const int row = blockIdx.x;
  const float* s = S + (size_t)row * SEQ;
  _Float16* p = P + (size_t)row * SEQ;
  __shared__ float red[256];
  const int tid = threadIdx.x;

  float m = -3.4e38f;
  for (int c = tid; c < SEQ; c += 256) m = fmaxf(m, s[c]);
  red[tid] = m;
  __syncthreads();
  for (int off = 128; off > 0; off >>= 1) {
    if (tid < off) red[tid] = fmaxf(red[tid], red[tid + off]);
    __syncthreads();
  }
  m = red[0];
  __syncthreads();

  float sum = 0.0f;
  for (int c = tid; c < SEQ; c += 256) sum += __expf(s[c] - m);
  red[tid] = sum;
  __syncthreads();
  for (int off = 128; off > 0; off >>= 1) {
    if (tid < off) red[tid] += red[tid + off];
    __syncthreads();
  }
  const float inv = 1.0f / red[0];
  for (int c = tid; c < SEQ; c += 256)
    p[c] = (_Float16)(__expf(s[c] - m) * inv);
}

// LDS-tiled V transpose: Vt[b][d][k] = QKV[(b*SEQ+k)*3072 + 2048 + d].
// 64x64 f16 tiles; reads and writes both coalesced (128B per 2-wave row).
__global__ void transpose_v_kernel(const _Float16* __restrict__ QKV,
                                   _Float16* __restrict__ Vt) {
  __shared__ _Float16 tile[64][65];   // +1 half pad to break bank conflicts
  const int k0 = blockIdx.x * 64;
  const int d0 = blockIdx.y * 64;
  const int b  = blockIdx.z;
  const int tx = threadIdx.x & 63;
  const int ty = threadIdx.x >> 6;    // 0..3

  const _Float16* src = QKV + (size_t)(b * SEQ + k0) * QKV_LD + 2 * DATT + d0;
#pragma unroll
  for (int r = 0; r < 64; r += 4)
    tile[r + ty][tx] = src[(size_t)(r + ty) * QKV_LD + tx];
  __syncthreads();

  _Float16* dst = Vt + ((size_t)b * DATT + d0) * SEQ + k0;
#pragma unroll
  for (int r = 0; r < 64; r += 4)
    dst[(size_t)(r + ty) * SEQ + tx] = tile[tx][r + ty];
}

// Context (one batch): Ctx[2048x1024] = P[2048x2048] * V[2048x1024] via Vt.
__global__ void ctx_gemm_kernel(const _Float16* __restrict__ P,
                                const _Float16* __restrict__ Vt,
                                _Float16* __restrict__ Ctx) {
  const int lane = threadIdx.x & 31;
  const int wave = threadIdx.x >> 5;
  const int m0 = blockIdx.x * 256 + wave * 32;
  const int n0 = blockIdx.y * 64;
  v8f accLo[4] = {}, accHi[4] = {};
  wave_gemm_32x64<SEQ / 32>(P, SEQ, m0, Vt, SEQ, n0, accLo, accHi, lane);
#pragma unroll
  for (int t = 0; t < 4; ++t) {
    store_d_f16(Ctx, DATT, m0, n0 + 16 * t, accLo[t], 1.0f, lane);
    store_d_f16(Ctx, DATT, m0 + 16, n0 + 16 * t, accHi[t], 1.0f, lane);
  }
}

// Output projection: Out[8192x128] = Ctx * Wout^T + bias (f32 out).
__global__ void out_gemm_kernel(const _Float16* __restrict__ Ctx,
                                const _Float16* __restrict__ Wo,
                                const float* __restrict__ bias,
                                float* __restrict__ Out) {
  const int lane = threadIdx.x & 31;
  const int wave = threadIdx.x >> 5;
  const int m0 = blockIdx.x * 256 + wave * 32;
  const int n0 = blockIdx.y * 64;
  v8f accLo[4] = {}, accHi[4] = {};
  wave_gemm_32x64<DATT / 32>(Ctx, DATT, m0, Wo, DATT, n0, accLo, accHi, lane);
  const int nl = lane & 15;
  const int mb = m0 + ((lane >> 4) << 3);
#pragma unroll
  for (int t = 0; t < 4; ++t) {
    const int n = n0 + 16 * t + nl;
    const float bv = bias[n];
#pragma unroll
    for (int r = 0; r < 8; ++r) {
      Out[(size_t)(mb + r) * DMODEL + n]      = accLo[t][r] + bv;
      Out[(size_t)(mb + 16 + r) * DMODEL + n] = accHi[t][r] + bv;
    }
  }
}

// ---------------------------------------------------------------------------
static inline size_t align256(size_t x) { return (x + 255) & ~(size_t)255; }

extern "C" void kernel_launch(void* const* d_in, const int* in_sizes, int n_in,
                              void* d_out, int out_size, void* d_ws, size_t ws_size,
                              hipStream_t stream) {
  const float* seq   = (const float*)d_in[0];  // [4,2048,128]
  const float* w_qkv = (const float*)d_in[1];  // [3072,128]
  const float* w_out = (const float*)d_in[2];  // [128,1024]
  const float* b_out = (const float*)d_in[3];  // [128]
  float* out = (float*)d_out;                  // [4,2048,128]

  char* ws = (char*)d_ws;
  size_t off = 0;
  _Float16* Xh   = (_Float16*)(ws + off); off += align256((size_t)M_ALL * DMODEL * 2);
  _Float16* Wh   = (_Float16*)(ws + off); off += align256((size_t)QKV_LD * DMODEL * 2);
  _Float16* Woh  = (_Float16*)(ws + off); off += align256((size_t)DMODEL * DATT * 2);
  _Float16* QKVh = (_Float16*)(ws + off); off += align256((size_t)M_ALL * QKV_LD * 2);
  _Float16* Vt   = (_Float16*)(ws + off); off += align256((size_t)BATCH * DATT * SEQ * 2);
  _Float16* Ctxh = (_Float16*)(ws + off); off += align256((size_t)M_ALL * DATT * 2);
  float*    Sb   = (float*)   (ws + off); off += align256((size_t)SEQ * SEQ * 4);  // per-batch reuse
  _Float16* Ph   = (_Float16*)(ws + off); off += align256((size_t)SEQ * SEQ * 2);  // per-batch reuse

  // 1) fp32 -> f16 conversions
  {
    int n;
    n = M_ALL * DMODEL;
    f32_to_f16_kernel<<<(n + 255) / 256, 256, 0, stream>>>(seq, Xh, n);
    n = QKV_LD * DMODEL;
    f32_to_f16_kernel<<<(n + 255) / 256, 256, 0, stream>>>(w_qkv, Wh, n);
    n = DMODEL * DATT;
    f32_to_f16_kernel<<<(n + 255) / 256, 256, 0, stream>>>(w_out, Woh, n);
  }

  // 2) QKV projection (Q pre-scaled by 1/32)
  qkv_gemm_kernel<<<dim3(M_ALL / 256, QKV_LD / 64), 256, 0, stream>>>(Xh, Wh, QKVh);

  // 3) V transpose (LDS-tiled, coalesced both sides)
  transpose_v_kernel<<<dim3(SEQ / 64, DATT / 64, BATCH), 256, 0, stream>>>(QKVh, Vt);

  // 4) Per-batch attention (score buffer reused across batches; same stream)
  for (int b = 0; b < BATCH; ++b) {
    const _Float16* Qb = QKVh + (size_t)b * SEQ * QKV_LD;
    const _Float16* Kb = Qb + DATT;
    scores_gemm_kernel<<<dim3(SEQ / 256, SEQ / 64), 256, 0, stream>>>(Qb, Kb, Sb);
    softmax_rows_kernel<<<SEQ, 256, 0, stream>>>(Sb, Ph);
    ctx_gemm_kernel<<<dim3(SEQ / 256, DATT / 64), 256, 0, stream>>>(
        Ph, Vt + (size_t)b * DATT * SEQ, Ctxh + (size_t)b * SEQ * DATT);
  }

  // 5) Output projection + bias
  out_gemm_kernel<<<dim3(M_ALL / 256, DMODEL / 64), 256, 0, stream>>>(Ctxh, Woh, b_out, out);
}
